// IGMMBayesianMLP_79001628443355
// MI455X (gfx1250) — matrix operand order repaired and checked by reading, not compiled, not run on hardware
//
#include <hip/hip_runtime.h>
#include <stdint.h>

typedef __attribute__((ext_vector_type(16))) __bf16 v16bf;
typedef __attribute__((ext_vector_type(8)))  float  v8f;

union FragU { v16bf v; uint4 q[2]; };

static __device__ __forceinline__ unsigned short f32_to_bf16(float f) {
    unsigned int u = __float_as_uint(f);
    unsigned int r = u + 0x7FFFu + ((u >> 16) & 1u);
    return (unsigned short)(r >> 16);
}

// ---------------------------------------------------------------------------
// Convert f32 -> bf16, 4 elements per thread (b128 load, b64 store)
// ---------------------------------------------------------------------------
__global__ __launch_bounds__(256) void cvt_kernel(const float* __restrict__ in,
                                                  unsigned short* __restrict__ out,
                                                  int n4) {
    int idx = blockIdx.x * blockDim.x + threadIdx.x;
    if (idx >= n4) return;
    size_t base = (size_t)idx * 4;
    float4 v = *(const float4*)(in + base);
    ushort4 o;
    o.x = f32_to_bf16(v.x); o.y = f32_to_bf16(v.y);
    o.z = f32_to_bf16(v.z); o.w = f32_to_bf16(v.w);
    *(ushort4*)(out + base) = o;
}

// ---------------------------------------------------------------------------
// W_eff[o,i] = sum_k softmax(ml)[k] * (w_mu[k,o,i] + exp(0.5*w_lv[k])*w_n[k,o,i])
// Output bf16 row-major [O, I]. 4 elements per thread.
// ---------------------------------------------------------------------------
__global__ __launch_bounds__(256) void reduce_w_kernel(const float* __restrict__ wmu,
                                                       const float* __restrict__ wn,
                                                       const float* __restrict__ wlv,
                                                       const float* __restrict__ ml,
                                                       unsigned short* __restrict__ wout,
                                                       int n4) {
    int idx = blockIdx.x * blockDim.x + threadIdx.x;
    if (idx >= n4) return;
    float l0 = ml[0], l1 = ml[1], l2 = ml[2], l3 = ml[3];
    float mx = fmaxf(fmaxf(l0, l1), fmaxf(l2, l3));
    float e0 = __expf(l0 - mx), e1 = __expf(l1 - mx), e2 = __expf(l2 - mx), e3 = __expf(l3 - mx);
    float inv = 1.0f / (e0 + e1 + e2 + e3);
    float mw[4] = {e0 * inv, e1 * inv, e2 * inv, e3 * inv};
    float sc[4];
#pragma unroll
    for (int k = 0; k < 4; ++k) sc[k] = __expf(0.5f * wlv[k]);

    size_t base   = (size_t)idx * 4;
    size_t stride = (size_t)n4 * 4;
    float ax = 0.f, ay = 0.f, az = 0.f, aw = 0.f;
#pragma unroll
    for (int k = 0; k < 4; ++k) {
        float4 a = *(const float4*)(wmu + k * stride + base);
        float4 b = *(const float4*)(wn  + k * stride + base);
        ax += mw[k] * (a.x + sc[k] * b.x);
        ay += mw[k] * (a.y + sc[k] * b.y);
        az += mw[k] * (a.z + sc[k] * b.z);
        aw += mw[k] * (a.w + sc[k] * b.w);
    }
    ushort4 o;
    o.x = f32_to_bf16(ax); o.y = f32_to_bf16(ay);
    o.z = f32_to_bf16(az); o.w = f32_to_bf16(aw);
    *(ushort4*)(wout + base) = o;
}

// ---------------------------------------------------------------------------
// b_eff[o] = sum_k softmax(ml)[k] * (b_mu[k,o] + exp(0.5*b_lv[k])*b_n[k,o]), f32
// ---------------------------------------------------------------------------
__global__ __launch_bounds__(256) void reduce_b_kernel(const float* __restrict__ bmu,
                                                       const float* __restrict__ bn,
                                                       const float* __restrict__ blv,
                                                       const float* __restrict__ ml,
                                                       float* __restrict__ bout,
                                                       int n) {
    int idx = blockIdx.x * blockDim.x + threadIdx.x;
    if (idx >= n) return;
    float l0 = ml[0], l1 = ml[1], l2 = ml[2], l3 = ml[3];
    float mx = fmaxf(fmaxf(l0, l1), fmaxf(l2, l3));
    float e0 = __expf(l0 - mx), e1 = __expf(l1 - mx), e2 = __expf(l2 - mx), e3 = __expf(l3 - mx);
    float inv = 1.0f / (e0 + e1 + e2 + e3);
    float mw[4] = {e0 * inv, e1 * inv, e2 * inv, e3 * inv};
    float acc = 0.f;
#pragma unroll
    for (int k = 0; k < 4; ++k)
        acc += mw[k] * (bmu[(size_t)k * n + idx] + __expf(0.5f * blv[k]) * bn[(size_t)k * n + idx]);
    bout[idx] = acc;
}

// ---------------------------------------------------------------------------
// GEMM: out[m,n] = act( sum_i A[m,i]*Wt[n,i] + bias[n] )
// A: [M,Kdim] bf16 row-major, Wt: [Ncols,Kdim] bf16 row-major.
// Block = 4 waves (128 thr); block tile 128(M) x 64(N); wave tile 32x64
// via 2x4 accumulators of v_wmma_f32_16x16x32_bf16.
// MODE 0: relu -> bf16 store (next layer input); MODE 1: raw f32 store.
// ---------------------------------------------------------------------------
template <int MODE>
__global__ __launch_bounds__(128) void gemm_bias_act_kernel(
    const unsigned short* __restrict__ A,
    const unsigned short* __restrict__ Wt,
    const float* __restrict__ bias,
    int Kdim, int Ncols,
    unsigned short* __restrict__ outBf,
    float* __restrict__ outF) {
    const int lane = threadIdx.x & 31;
    const int wave = threadIdx.x >> 5;
    const int hh   = lane >> 4;   // lane half: 0 or 1
    const int l16  = lane & 15;

    const int mBase = blockIdx.x * 128 + wave * 32;
    const int nBase = blockIdx.y * 64;

    // Init accumulators with the bias (depends only on the N column = l16 slot).
    v8f acc[2][4];
#pragma unroll
    for (int ns = 0; ns < 4; ++ns) {
        float bv = bias[nBase + ns * 16 + l16];
        v8f c;
#pragma unroll
        for (int e = 0; e < 8; ++e) c[e] = bv;
        acc[0][ns] = c;
        acc[1][ns] = c;
    }

    // A fragment rows (16-bit A 16x32 layout): lane holds row m = l16,
    // chunk0 = K[hh*8 .. +7], chunk1 = K[16+hh*8 .. +7]  (two b128 loads).
    const unsigned short* aRow[2];
    aRow[0] = A + (size_t)(mBase + l16) * Kdim;
    aRow[1] = aRow[0] + (size_t)16 * Kdim;
    // B fragment (32x16): lane holds column n = l16, chunk0 = K[hh*8 .. +7], chunk1 = K[16+hh*8 .. +7].
    const unsigned short* wRow[4];
#pragma unroll
    for (int ns = 0; ns < 4; ++ns)
        wRow[ns] = Wt + (size_t)(nBase + ns * 16 + l16) * Kdim;

    const int aOff0 = hh * 8;
    const int aOff1 = 16 + hh * 8;
    const int bOff  = hh * 8;

    for (int k = 0; k < Kdim; k += 32) {
        FragU aF[2], bF[4];
#pragma unroll
        for (int ms = 0; ms < 2; ++ms) {
            aF[ms].q[0] = *(const uint4*)(aRow[ms] + k + aOff0);
            aF[ms].q[1] = *(const uint4*)(aRow[ms] + k + aOff1);
        }
#pragma unroll
        for (int ns = 0; ns < 4; ++ns) {
            bF[ns].q[0] = *(const uint4*)(wRow[ns] + k + bOff);
            bF[ns].q[1] = *(const uint4*)(wRow[ns] + k + bOff + 16);
        }
#pragma unroll
        for (int ms = 0; ms < 2; ++ms)
#pragma unroll
            for (int ns = 0; ns < 4; ++ns)
                acc[ms][ns] = __builtin_amdgcn_wmma_f32_16x16x32_bf16(
                    false, aF[ms].v, false, bF[ns].v, (short)0, acc[ms][ns],
                    false, false);
    }

    // C/D layout: element e of v8f -> row = hh*8 + e, col = l16 (within 16x16 tile).
#pragma unroll
    for (int ms = 0; ms < 2; ++ms) {
#pragma unroll
        for (int ns = 0; ns < 4; ++ns) {
            const int col = nBase + ns * 16 + l16;
#pragma unroll
            for (int e = 0; e < 8; ++e) {
                const int row = mBase + ms * 16 + hh * 8 + e;
                float v = acc[ms][ns][e];
                if (MODE == 0) {
                    v = v > 0.f ? v : 0.f;
                    outBf[(size_t)row * Ncols + col] = f32_to_bf16(v);
                } else {
                    outF[(size_t)row * Ncols + col] = v;
                }
            }
        }
    }
}

// ---------------------------------------------------------------------------
// In-place row softmax over `cols` f32 values. One 256-thread block per row.
// ---------------------------------------------------------------------------
__global__ __launch_bounds__(256) void softmax_rows_kernel(float* __restrict__ data, int cols) {
    __shared__ float red[256];
    const int row = blockIdx.x;
    float* p = data + (size_t)row * cols;
    const int t = threadIdx.x;

    float m = -3.402823466e+38f;
    for (int c = t; c < cols; c += 256) m = fmaxf(m, p[c]);
    red[t] = m; __syncthreads();
    for (int s = 128; s > 0; s >>= 1) {
        if (t < s) red[t] = fmaxf(red[t], red[t + s]);
        __syncthreads();
    }
    m = red[0]; __syncthreads();

    float sum = 0.f;
    for (int c = t; c < cols; c += 256) {
        float e = __expf(p[c] - m);
        p[c] = e;
        sum += e;
    }
    red[t] = sum; __syncthreads();
    for (int s = 128; s > 0; s >>= 1) {
        if (t < s) red[t] += red[t + s];
        __syncthreads();
    }
    float inv = 1.0f / red[0];
    for (int c = t; c < cols; c += 256) p[c] *= inv;
}

// ---------------------------------------------------------------------------
extern "C" void kernel_launch(void* const* d_in, const int* in_sizes, int n_in,
                              void* d_out, int out_size, void* d_ws, size_t ws_size,
                              hipStream_t stream) {
    (void)in_sizes; (void)n_in; (void)out_size; (void)ws_size;
    constexpr int Bn = 8192, DIN = 2048, Hd = 2048, DOUT = 1024;

    const float* x = (const float*)d_in[0];
    const float* w_mu[3] = {(const float*)d_in[1],  (const float*)d_in[8],  (const float*)d_in[15]};
    const float* b_mu[3] = {(const float*)d_in[2],  (const float*)d_in[9],  (const float*)d_in[16]};
    const float* w_lv[3] = {(const float*)d_in[3],  (const float*)d_in[10], (const float*)d_in[17]};
    const float* b_lv[3] = {(const float*)d_in[4],  (const float*)d_in[11], (const float*)d_in[18]};
    const float* ml[3]   = {(const float*)d_in[5],  (const float*)d_in[12], (const float*)d_in[19]};
    const float* w_n[3]  = {(const float*)d_in[6],  (const float*)d_in[13], (const float*)d_in[20]};
    const float* b_n[3]  = {(const float*)d_in[7],  (const float*)d_in[14], (const float*)d_in[21]};

    char* ws = (char*)d_ws;
    size_t off = 0;
    auto alloc = [&](size_t bytes) -> char* {
        char* p = ws + off;
        off += (bytes + 255) & ~(size_t)255;
        return p;
    };
    unsigned short* xb = (unsigned short*)alloc((size_t)Bn * DIN * 2);
    unsigned short* h1 = (unsigned short*)alloc((size_t)Bn * Hd * 2);
    unsigned short* h2 = (unsigned short*)alloc((size_t)Bn * Hd * 2);
    unsigned short* W0 = (unsigned short*)alloc((size_t)Hd * DIN * 2);
    unsigned short* W1 = (unsigned short*)alloc((size_t)Hd * Hd * 2);
    unsigned short* W2 = (unsigned short*)alloc((size_t)DOUT * Hd * 2);
    float* bb0 = (float*)alloc((size_t)Hd * 4);
    float* bb1 = (float*)alloc((size_t)Hd * 4);
    float* bb2 = (float*)alloc((size_t)DOUT * 4);

    // x -> bf16
    {
        int n4 = Bn * DIN / 4;
        cvt_kernel<<<n4 / 256, 256, 0, stream>>>(x, xb, n4);
    }
    // Effective (mixture-collapsed) weights and biases
    reduce_w_kernel<<<(Hd * DIN / 4) / 256, 256, 0, stream>>>(w_mu[0], w_n[0], w_lv[0], ml[0], W0, Hd * DIN / 4);
    reduce_w_kernel<<<(Hd * Hd  / 4) / 256, 256, 0, stream>>>(w_mu[1], w_n[1], w_lv[1], ml[1], W1, Hd * Hd / 4);
    reduce_w_kernel<<<(DOUT * Hd / 4) / 256, 256, 0, stream>>>(w_mu[2], w_n[2], w_lv[2], ml[2], W2, DOUT * Hd / 4);
    reduce_b_kernel<<<Hd / 256,   256, 0, stream>>>(b_mu[0], b_n[0], b_lv[0], ml[0], bb0, Hd);
    reduce_b_kernel<<<Hd / 256,   256, 0, stream>>>(b_mu[1], b_n[1], b_lv[1], ml[1], bb1, Hd);
    reduce_b_kernel<<<DOUT / 256, 256, 0, stream>>>(b_mu[2], b_n[2], b_lv[2], ml[2], bb2, DOUT);

    // Three WMMA GEMMs, relu fused into layers 0/1, logits f32 into d_out
    dim3 blk(128);
    gemm_bias_act_kernel<0><<<dim3(Bn / 128, Hd / 64),   blk, 0, stream>>>(xb, W0, bb0, DIN, Hd,  h1, nullptr);
    gemm_bias_act_kernel<0><<<dim3(Bn / 128, Hd / 64),   blk, 0, stream>>>(h1, W1, bb1, Hd,  Hd,  h2, nullptr);
    gemm_bias_act_kernel<1><<<dim3(Bn / 128, DOUT / 64), blk, 0, stream>>>(h2, W2, bb2, Hd,  DOUT, nullptr, (float*)d_out);

    // Row softmax in-place in d_out
    softmax_rows_kernel<<<Bn, 256, 0, stream>>>((float*)d_out, DOUT);
}
